// TemporalAttention_3590592659669
// MI455X (gfx1250) — matrix operand order repaired and checked by reading, not compiled, not run on hardware
//
#include <hip/hip_runtime.h>
#include <math.h>
#include <stdint.h>

// ---------- problem constants ----------
#define KB   128          // batch
#define KT   192          // time
#define KE   1024         // embed
#define KEH  512          // E/2 and head_dim
#define KROWS (KB*KT)     // 24576

typedef __bf16 bf16_t;
typedef bf16_t v16bf __attribute__((ext_vector_type(16)));
typedef float  v8f   __attribute__((ext_vector_type(8)));

union FragAB { unsigned u[8]; uint4 q[2]; v16bf v; };

__device__ __forceinline__ unsigned short f2bf(float f) {
    union { float f; unsigned u; } x; x.f = f;
    unsigned r = (x.u + 0x7FFFu + ((x.u >> 16) & 1u)) >> 16;   // RNE
    return (unsigned short)r;
}
__device__ __forceinline__ unsigned f2bf_pk(float lo, float hi) {
    return (unsigned)f2bf(lo) | ((unsigned)f2bf(hi) << 16);
}
__device__ __forceinline__ float gelu_exact(float x) {
    return 0.5f * x * (1.0f + erff(x * 0.70710678118654752f));
}

// ---------- convert imp_w1 to bf16 ----------
__global__ __launch_bounds__(256) void k_cvt(const float* __restrict__ src,
                                             unsigned short* __restrict__ dst, int n) {
    int i = blockIdx.x * 256 + threadIdx.x;
    if (i < n) dst[i] = f2bf(src[i]);
}

// ---------- importance logits: logits[row] = b2 + sum_j w2[j]*gelu(X@W1^T + b1)[row,j]
// One workgroup per 16-row M-tile. 8 waves; wave w covers N=[64w,64w+64) of N=512.
__global__ __launch_bounds__(256) void k_imp(const float* __restrict__ x,
                                             const unsigned short* __restrict__ w1bf,
                                             const float* __restrict__ b1,
                                             const float* __restrict__ w2,
                                             const float* __restrict__ b2,
                                             float* __restrict__ logits) {
    __shared__ __align__(16) unsigned short Atile[16 * KE];   // 32 KB bf16
    __shared__ float red[16];

    const int tid   = threadIdx.x;
    const int mbase = blockIdx.x * 16;

    // Stage 16 rows of x as bf16 into LDS (coalesced f32 reads).
    for (int i = tid; i < 16 * KE; i += 256) {
        int r = i >> 10, c = i & (KE - 1);
        Atile[i] = f2bf(x[(size_t)(mbase + r) * KE + c]);
    }
    if (tid < 16) red[tid] = 0.0f;
    __syncthreads();

    const int lane = tid & 31;
    const int wave = tid >> 5;
    const int l16  = lane & 15;
    const int g    = lane >> 4;
    const int nbase = wave * 64;

    v8f acc[4] = {};
    for (int ks = 0; ks < KE; ks += 32) {
        // A fragment: lane row = l16, k pairs contiguous from ks+8g and ks+16+8g
        FragAB a;
        a.q[0] = *(const uint4*)(Atile + (size_t)l16 * KE + ks + g * 8);
        a.q[1] = *(const uint4*)(Atile + (size_t)l16 * KE + ks + 16 + g * 8);
        #pragma unroll
        for (int nt = 0; nt < 4; nt++) {
            const int n = nbase + nt * 16 + l16;
            FragAB b;
            const unsigned short* bp = w1bf + (size_t)n * KE + ks + g * 8;
            b.q[0] = *(const uint4*)(bp);
            b.q[1] = *(const uint4*)(bp + 16);
            acc[nt] = __builtin_amdgcn_wmma_f32_16x16x32_bf16(
                false, a.v, false, b.v, (short)0, acc[nt], false, false);
        }
    }

    // Fused epilogue: gelu + dot with w2, reduce over N.
    float wsum[8] = {0.f,0.f,0.f,0.f,0.f,0.f,0.f,0.f};
    #pragma unroll
    for (int nt = 0; nt < 4; nt++) {
        const int n = nbase + nt * 16 + l16;
        const float w2v = w2[n], b1v = b1[n];
        #pragma unroll
        for (int r = 0; r < 8; r++) {
            wsum[r] += w2v * gelu_exact(acc[nt][r] + b1v);
        }
    }
    #pragma unroll
    for (int r = 0; r < 8; r++) {
        float v = wsum[r];
        v += __shfl_xor(v, 1, 32);
        v += __shfl_xor(v, 2, 32);
        v += __shfl_xor(v, 4, 32);
        v += __shfl_xor(v, 8, 32);
        if (l16 == 0) atomicAdd(&red[g * 8 + r], v);   // ds_add_f32
    }
    __syncthreads();
    if (tid < 16) logits[mbase + tid] = red[tid] + b2[0];
}

// ---------- softmax over T per batch ----------
__global__ __launch_bounds__(256) void k_softmax(const float* __restrict__ logits,
                                                 float* __restrict__ iw) {
    __shared__ float sh[256];
    const int b = blockIdx.x, tid = threadIdx.x;
    float v = (tid < KT) ? logits[b * KT + tid] : -INFINITY;
    sh[tid] = v; __syncthreads();
    for (int s = 128; s > 0; s >>= 1) { if (tid < s) sh[tid] = fmaxf(sh[tid], sh[tid + s]); __syncthreads(); }
    const float mx = sh[0]; __syncthreads();
    float e = (tid < KT) ? expf(v - mx) : 0.0f;
    sh[tid] = e; __syncthreads();
    for (int s = 128; s > 0; s >>= 1) { if (tid < s) sh[tid] += sh[tid + s]; __syncthreads(); }
    const float inv = 1.0f / sh[0];
    if (tid < KT) iw[b * KT + tid] = e * inv;
}

// ---------- final_output[b,e] = sum_t x[b,t,e] * iw[b,t] ----------
__global__ __launch_bounds__(256) void k_weighted(const float* __restrict__ x,
                                                  const float* __restrict__ iw,
                                                  float* __restrict__ fw) {
    const int b = blockIdx.y;
    const int e = blockIdx.x * 256 + threadIdx.x;
    float s = 0.0f;
    for (int t = 0; t < KT; t++)
        s += x[((size_t)b * KT + t) * KE + e] * iw[b * KT + t];
    fw[(size_t)b * KE + e] = s;
}

// ---------- generic M=128 WMMA GEMM: C = A(128xK) * op(W) + bias
// BT=1: W stored (N,K) row-major (C=A@W^T). BT=0: W stored (K,N) row-major.
// grid.x = N/16 (one 16-col tile per WG); 8 waves = 8 row tiles of 16.
template <int BT>
__global__ __launch_bounds__(256) void k_gemm(const float* __restrict__ A, int64_t lda,
                                              const float* __restrict__ W, int64_t ldw,
                                              const float* __restrict__ bias,
                                              float* __restrict__ C, int64_t ldc, int K) {
    const int tid  = threadIdx.x;
    const int lane = tid & 31;
    const int wave = tid >> 5;
    const int l16  = lane & 15;
    const int g    = lane >> 4;
    const int n    = blockIdx.x * 16 + l16;
    const int mrow = wave * 16 + l16;

    v8f acc = {};
    for (int ks = 0; ks < K; ks += 32) {
        FragAB a, b;
        const float* ar = A + (size_t)mrow * lda + ks;
        #pragma unroll
        for (int v = 0; v < 8; v++) {
            const int k = (v < 4) ? (g * 8 + 2 * v) : (16 + g * 8 + 2 * (v - 4));
            a.u[v] = f2bf_pk(ar[k], ar[k + 1]);
        }
        #pragma unroll
        for (int v = 0; v < 8; v++) {
            const int k = ks + ((v < 4) ? (g * 8 + 2 * v) : (16 + g * 8 + 2 * (v - 4)));
            float lo, hi;
            if (BT) { const float* wr = W + (size_t)n * ldw + k; lo = wr[0]; hi = wr[1]; }
            else    { lo = W[(size_t)k * ldw + n]; hi = W[(size_t)(k + 1) * ldw + n]; }
            b.u[v] = f2bf_pk(lo, hi);
        }
        acc = __builtin_amdgcn_wmma_f32_16x16x32_bf16(
            false, a.v, false, b.v, (short)0, acc, false, false);
    }
    const float bv = bias ? bias[n] : 0.0f;
    #pragma unroll
    for (int r = 0; r < 8; r++) {
        const int m = wave * 16 + r + 8 * g;
        C[(size_t)m * ldc + n] = acc[r] + bv;
    }
}

// ---------- attention core per (b, head): scores via qk·x rows, softmax, context
__global__ __launch_bounds__(256) void k_attn(const float* __restrict__ x,
                                              const float* __restrict__ qk,
                                              float* __restrict__ ctx,
                                              int S, int t0, int tstep) {
    __shared__ float sc[64];
    __shared__ float at[64];
    const int b = blockIdx.x, h = blockIdx.y;
    const int tid = threadIdx.x, lane = tid & 31, wave = tid >> 5;
    const float* qkr = qk + ((size_t)h * KB + b) * KE;

    for (int k = wave; k < S; k += 8) {
        const float* xr = x + ((size_t)b * KT + (t0 + k * tstep)) * KE;
        float d = 0.0f;
        for (int e = lane; e < KE; e += 32) d += qkr[e] * xr[e];
        d += __shfl_xor(d, 16, 32);
        d += __shfl_xor(d, 8, 32);
        d += __shfl_xor(d, 4, 32);
        d += __shfl_xor(d, 2, 32);
        d += __shfl_xor(d, 1, 32);
        if (lane == 0) sc[k] = d * 0.04419417382415922f;   // 1/sqrt(512)
    }
    __syncthreads();
    float mx = -INFINITY;
    for (int k = 0; k < S; k++) mx = fmaxf(mx, sc[k]);
    float sum = 0.0f;
    for (int k = 0; k < S; k++) sum += expf(sc[k] - mx);
    const float inv = 1.0f / sum;
    if (tid < S) at[tid] = expf(sc[tid] - mx) * inv;
    __syncthreads();

    for (int e = tid; e < KE; e += 256) {
        float a = 0.0f;
        for (int k = 0; k < S; k++)
            a += at[k] * x[((size_t)b * KT + (t0 + k * tstep)) * KE + e];
        ctx[((size_t)h * KB + b) * KE + e] = a;
    }
}

// ---------- out = 0.5*combined + 0.5*finalw ----------
__global__ __launch_bounds__(256) void k_combine(const float* __restrict__ comb,
                                                 const float* __restrict__ fw,
                                                 float* __restrict__ out, int n) {
    int i = blockIdx.x * 256 + threadIdx.x;
    if (i < n) out[i] = 0.5f * (comb[i] + fw[i]);
}

extern "C" void kernel_launch(void* const* d_in, const int* in_sizes, int n_in,
                              void* d_out, int out_size, void* d_ws, size_t ws_size,
                              hipStream_t stream) {
    (void)in_sizes; (void)n_in; (void)out_size; (void)ws_size;

    const float* x       = (const float*)d_in[0];
    const float* in_w[3] = {(const float*)d_in[1], (const float*)d_in[5], (const float*)d_in[9]};
    const float* in_b[3] = {(const float*)d_in[2], (const float*)d_in[6], (const float*)d_in[10]};
    const float* out_w[3]= {(const float*)d_in[3], (const float*)d_in[7], (const float*)d_in[11]};
    const float* out_b[3]= {(const float*)d_in[4], (const float*)d_in[8], (const float*)d_in[12]};
    const float* mixer_w = (const float*)d_in[13];
    const float* mixer_b = (const float*)d_in[14];
    const float* imp_w1  = (const float*)d_in[15];
    const float* imp_b1  = (const float*)d_in[16];
    const float* imp_w2  = (const float*)d_in[17];
    const float* imp_b2  = (const float*)d_in[18];

    float* out = (float*)d_out;          // (B,E)
    float* iw  = out + (size_t)KB * KE;  // (B,T)

    // workspace layout (w1bf first for 16B alignment)
    char* p = (char*)d_ws;
    unsigned short* w1bf = (unsigned short*)p;  p += (size_t)KEH * KE * sizeof(unsigned short);
    float* logits = (float*)p;                  p += (size_t)KROWS * sizeof(float);
    float* fw     = (float*)p;                  p += (size_t)KB * KE * sizeof(float);
    float* qp     = (float*)p;                  p += (size_t)KB * KE * sizeof(float);
    float* qk     = (float*)p;                  p += (size_t)2 * KB * KE * sizeof(float);
    float* ctx    = (float*)p;                  p += (size_t)2 * KB * KE * sizeof(float);
    float* obuf   = (float*)p;                  p += (size_t)KB * KE * sizeof(float);
    float* ms     = (float*)p;                  p += (size_t)KB * 3 * KE * sizeof(float);
    float* comb   = (float*)p;                  p += (size_t)KB * KE * sizeof(float);

    // importance path
    k_cvt<<<(KEH * KE) / 256, 256, 0, stream>>>(imp_w1, w1bf, KEH * KE);
    k_imp<<<KROWS / 16, 256, 0, stream>>>(x, w1bf, imp_b1, imp_w2, imp_b2, logits);
    k_softmax<<<KB, 256, 0, stream>>>(logits, iw);
    k_weighted<<<dim3(KE / 256, KB), 256, 0, stream>>>(x, iw, fw);

    // multi-scale single-query attention (restructured: no explicit K/V projections)
    const int  Ss[3]    = {3, 3, 64};
    const int  t0s[3]   = {KT - 3, KT - 6, 0};
    const int  steps[3] = {1, 2, 3};
    for (int s = 0; s < 3; s++) {
        // qp = q @ wq^T + bq          (128x1024) * (1024x1024)^T
        k_gemm<1><<<KE / 16, 256, 0, stream>>>(x + (size_t)(KT - 1) * KE, (int64_t)KT * KE,
                                               in_w[s], KE, in_b[s], qp, KE, KE);
        // qk_h = qp_h @ wk_h          (128x512) * (512x1024)
        for (int h = 0; h < 2; h++) {
            k_gemm<0><<<KE / 16, 256, 0, stream>>>(qp + h * KEH, KE,
                                                   in_w[s] + (size_t)(KE + h * KEH) * KE, KE,
                                                   nullptr, qk + (size_t)h * KB * KE, KE, KEH);
        }
        // scores / softmax / context
        k_attn<<<dim3(KB, 2), 256, 0, stream>>>(x, qk, ctx, Ss[s], t0s[s], steps[s]);
        // o_h = ctx_h @ wv_h^T + bv_h (128x1024) * (512x1024)^T
        for (int h = 0; h < 2; h++) {
            k_gemm<1><<<KEH / 16, 256, 0, stream>>>(ctx + (size_t)h * KB * KE, KE,
                                                    in_w[s] + (size_t)(2 * KE + h * KEH) * KE, KE,
                                                    in_b[s] + 2 * KE + h * KEH,
                                                    obuf + h * KEH, KE, KE);
        }
        // a_s = o @ out_w^T + out_b -> ms[:, s*E : (s+1)*E]
        k_gemm<1><<<KE / 16, 256, 0, stream>>>(obuf, KE, out_w[s], KE, out_b[s],
                                               ms + (size_t)s * KE, (int64_t)3 * KE, KE);
    }

    // combined = ms @ mixer_w^T + mixer_b   (128x3072) * (1024x3072)^T
    k_gemm<1><<<KE / 16, 256, 0, stream>>>(ms, (int64_t)3 * KE, mixer_w, (int64_t)3 * KE,
                                           mixer_b, comb, KE, 3 * KE);
    // out = 0.5*combined + 0.5*final
    k_combine<<<(KB * KE) / 256, 256, 0, stream>>>(comb, fw, out, KB * KE);
}
